// AttentionFlow_77859167142268
// MI455X (gfx1250) — matrix operand order repaired and checked by reading, not compiled, not run on hardware
//
#include <hip/hip_runtime.h>
#include <hip/hip_bf16.h>

typedef __attribute__((ext_vector_type(16))) __bf16 v16bf;
typedef __attribute__((ext_vector_type(8)))  float  v8f;

#define B_   8
#define T_   2048
#define Q_   512
#define D_   512

__device__ inline float wredmax(float v) {
#pragma unroll
  for (int o = 16; o; o >>= 1) v = fmaxf(v, __shfl_xor(v, o, 32));
  return v;
}
__device__ inline float wredsum(float v) {
#pragma unroll
  for (int o = 16; o; o >>= 1) v += __shfl_xor(v, o, 32);
  return v;
}
__device__ inline float4 mul4(float4 a, float4 b) {
  return make_float4(a.x * b.x, a.y * b.y, a.z * b.z, a.w * b.w);
}
template <int BASE>
__device__ inline void put8(v16bf& v, float4 x, float4 y) {
  v[BASE + 0] = (__bf16)x.x; v[BASE + 1] = (__bf16)x.y;
  v[BASE + 2] = (__bf16)x.z; v[BASE + 3] = (__bf16)x.w;
  v[BASE + 4] = (__bf16)y.x; v[BASE + 5] = (__bf16)y.y;
  v[BASE + 6] = (__bf16)y.z; v[BASE + 7] = (__bf16)y.w;
}

// ---------------- Kernel 1: sq[b,i] = <q[b,i], w_q> ----------------
__global__ __launch_bounds__(256) void sq_kernel(const float* __restrict__ q,
                                                 const float* __restrict__ w_q,
                                                 float* __restrict__ sq) {
  int row  = blockIdx.x * 8 + (threadIdx.x >> 5);  // [0, B_*Q_)
  int lane = threadIdx.x & 31;
  const float* qr = q + (size_t)row * D_;
  float acc = 0.f;
#pragma unroll
  for (int d = 0; d < 4; d++) {
    int off = lane * 4 + d * 128;
    float4 v = *(const float4*)(qr + off);
    float4 w = *(const float4*)(w_q + off);
    acc += v.x * w.x + v.y * w.y + v.z * w.z + v.w * w.w;
  }
  acc = wredsum(acc);
  if (lane == 0) sq[row] = acc;
}

// ---------------- Kernel 2: fused scq-GEMM + softmax + c2q-GEMM ----------------
// grid (T_/16, B_), 256 threads (8 waves). Wave w owns M-block 0 and N-blocks w*4..w*4+3.
__global__ __launch_bounds__(256) void attn_main(const float* __restrict__ c,
                                                 const float* __restrict__ q,
                                                 const float* __restrict__ w_c,
                                                 const float* __restrict__ w_cq,
                                                 const float* __restrict__ sq_ws,
                                                 float* __restrict__ m_ws,
                                                 float* __restrict__ out) {
  __shared__ __align__(16) float s_lds[16][Q_];   // 32 KB: scores -> probs
  __shared__ __align__(16) float wcq_lds[D_];
  __shared__ __align__(16) float wc_lds[D_];

  const int b    = blockIdx.y;
  const int t0   = blockIdx.x * 16;
  const int tid  = threadIdx.x;
  const int wid  = tid >> 5;
  const int lane = tid & 31;
  const int half = lane >> 4;   // 0: lanes 0-15, 1: lanes 16-31
  const int l16  = lane & 15;

  for (int i = tid; i < D_; i += 256) { wcq_lds[i] = w_cq[i]; wc_lds[i] = w_c[i]; }
  __syncthreads();

  const float* qbase = q + (size_t)b * Q_ * D_;
  const float* crow  = c + ((size_t)(b * T_ + t0) + l16) * D_;  // A row = t0 + l16

  // ---- GEMM1: s[t,i] = sum_d (c[t,d]*w_cq[d]) * q[i,d] ----
  v8f acc[4];
#pragma unroll
  for (int j = 0; j < 4; j++)
#pragma unroll
    for (int r = 0; r < 8; r++) acc[j][r] = 0.f;

  for (int k0 = 0; k0 < D_; k0 += 32) {
    // A fragment 16x32 bf16: K = e + 8*((e>=8) + half)
    float4 a0 = *(const float4*)(crow + k0 + 8 * half);
    float4 a1 = *(const float4*)(crow + k0 + 8 * half + 4);
    float4 a2 = *(const float4*)(crow + k0 + 16 + 8 * half);
    float4 a3 = *(const float4*)(crow + k0 + 16 + 8 * half + 4);
    float4 w0 = *(const float4*)(&wcq_lds[k0 + 8 * half]);
    float4 w1 = *(const float4*)(&wcq_lds[k0 + 8 * half + 4]);
    float4 w2 = *(const float4*)(&wcq_lds[k0 + 16 + 8 * half]);
    float4 w3 = *(const float4*)(&wcq_lds[k0 + 16 + 8 * half + 4]);
    v16bf A;
    put8<0>(A, mul4(a0, w0), mul4(a1, w1));
    put8<8>(A, mul4(a2, w2), mul4(a3, w3));
#pragma unroll
    for (int j = 0; j < 4; j++) {
      // B fragment 32x16 bf16: B[k][n] = q[i0+n][k0+k], K = e + 8*((e>=8) + half)
      int i0 = (wid * 4 + j) * 16;
      const float* qr = qbase + (size_t)(i0 + l16) * D_ + k0 + 8 * half;
      float4 b0 = *(const float4*)(qr);
      float4 b1 = *(const float4*)(qr + 4);
      float4 b2 = *(const float4*)(qr + 16);
      float4 b3 = *(const float4*)(qr + 20);
      v16bf Bv;
      put8<0>(Bv, b0, b1);
      put8<8>(Bv, b2, b3);
      acc[j] = __builtin_amdgcn_wmma_f32_16x16x32_bf16(false, A, false, Bv,
                                                       (short)0, acc[j], false, false);
    }
  }
  // epilogue: add sq[i], stage scores into LDS (C layout: M = r + 8*half, N = l16)
#pragma unroll
  for (int j = 0; j < 4; j++) {
    int i0 = (wid * 4 + j) * 16;
    float sqv = sq_ws[b * Q_ + i0 + l16];
#pragma unroll
    for (int r = 0; r < 8; r++) s_lds[r + 8 * half][i0 + l16] = acc[j][r] + sqv;
  }
  __syncthreads();

  // ---- softmax over i (per row), plus m[t] = rowmax + <c_t, w_c> ----
  for (int rr = 0; rr < 2; rr++) {
    int row = wid + rr * 8;
    float mx = -3.0e38f;
    for (int i = lane; i < Q_; i += 32) mx = fmaxf(mx, s_lds[row][i]);
    mx = wredmax(mx);
    float sum = 0.f;
    for (int i = lane; i < Q_; i += 32) {
      float p = __expf(s_lds[row][i] - mx);
      s_lds[row][i] = p;
      sum += p;
    }
    sum = wredsum(sum);
    float inv = 1.f / sum;
    for (int i = lane; i < Q_; i += 32) s_lds[row][i] *= inv;
    const float* cr = c + (size_t)(b * T_ + t0 + row) * D_;
    float sc = 0.f;
    for (int i = lane; i < D_; i += 32) sc += cr[i] * wc_lds[i];
    sc = wredsum(sc);
    if (lane == 0) m_ws[b * T_ + t0 + row] = mx + sc;
  }
  __syncthreads();

  // ---- GEMM2: c2q[t,d] = sum_i a[t,i] * q[i,d] ----
  v8f acc2[4];
#pragma unroll
  for (int j = 0; j < 4; j++)
#pragma unroll
    for (int r = 0; r < 8; r++) acc2[j][r] = 0.f;

  const float* sr = &s_lds[l16][0];  // A row = l16
  for (int k0 = 0; k0 < Q_; k0 += 32) {
    float4 x0 = *(const float4*)(sr + k0 + 8 * half);
    float4 x1 = *(const float4*)(sr + k0 + 8 * half + 4);
    float4 x2 = *(const float4*)(sr + k0 + 16 + 8 * half);
    float4 x3 = *(const float4*)(sr + k0 + 16 + 8 * half + 4);
    v16bf A;
    put8<0>(A, x0, x1);
    put8<8>(A, x2, x3);
#pragma unroll
    for (int j = 0; j < 4; j++) {
      // B[k][n] = q[k0+k][d0+n] (row-major natural), K = e + 8*((e>=8) + half)
      int d0 = (wid * 4 + j) * 16;
      const float* qcol = qbase + (size_t)(k0 + 8 * half) * D_ + d0 + l16;
      v16bf Bv;
#pragma unroll
      for (int e = 0; e < 16; e++) Bv[e] = (__bf16)qcol[(size_t)(e + 8 * (e >= 8)) * D_];
      acc2[j] = __builtin_amdgcn_wmma_f32_16x16x32_bf16(false, A, false, Bv,
                                                        (short)0, acc2[j], false, false);
    }
  }

  // ---- write out[..,512:1024]=c2q, out[..,1024:1536]=c*c2q ----
#pragma unroll
  for (int j = 0; j < 4; j++) {
    int d = (wid * 4 + j) * 16 + l16;
#pragma unroll
    for (int r = 0; r < 8; r++) {
      int row = r + 8 * half;
      size_t cidx  = (size_t)(b * T_ + t0 + row) * D_ + d;
      size_t obase = (size_t)(b * T_ + t0 + row) * 2048;
      float c2q = acc2[j][r];
      float cv  = c[cidx];
      out[obase + 512 + d]  = c2q;
      out[obase + 1024 + d] = cv * c2q;
    }
  }
  // ---- out[..,0:512] = c (vectorized copy) ----
  for (int idx = tid; idx < 16 * 128; idx += 256) {
    int row = idx >> 7;
    int col = (idx & 127) << 2;
    float4 v = *(const float4*)(c + (size_t)(b * T_ + t0 + row) * D_ + col);
    *(float4*)(out + (size_t)(b * T_ + t0 + row) * 2048 + col) = v;
  }
}

// ---------------- Kernel 3: beta = softmax_t(m[b,:]) ----------------
__global__ __launch_bounds__(256) void beta_kernel(const float* __restrict__ m,
                                                   float* __restrict__ beta) {
  int b = blockIdx.x;
  int tid = threadIdx.x, lane = tid & 31, wid = tid >> 5;
  __shared__ float red[8];
  const float* mb = m + b * T_;
  float mx = -3.0e38f;
  for (int t = tid; t < T_; t += 256) mx = fmaxf(mx, mb[t]);
  mx = wredmax(mx);
  if (lane == 0) red[wid] = mx;
  __syncthreads();
  float bm = red[0];
#pragma unroll
  for (int i = 1; i < 8; i++) bm = fmaxf(bm, red[i]);
  __syncthreads();
  float sum = 0.f;
  for (int t = tid; t < T_; t += 256) sum += __expf(mb[t] - bm);
  sum = wredsum(sum);
  if (lane == 0) red[wid] = sum;
  __syncthreads();
  float bs = 0.f;
#pragma unroll
  for (int i = 0; i < 8; i++) bs += red[i];
  float inv = 1.f / bs;
  for (int t = tid; t < T_; t += 256) beta[b * T_ + t] = __expf(mb[t] - bm) * inv;
}

// ---------------- Kernel 4: q2c[b,d] += sum_t beta[b,t]*c[b,t,d] ----------------
__global__ __launch_bounds__(256) void q2c_kernel(const float* __restrict__ c,
                                                  const float* __restrict__ beta,
                                                  float* __restrict__ q2c) {
  int b = blockIdx.y;
  int t0 = blockIdx.x * 128;
  int d = threadIdx.x * 2;
  float ax = 0.f, ay = 0.f;
  for (int t = 0; t < 128; t++) {
    float bt = beta[b * T_ + t0 + t];
    float2 cv = *(const float2*)(c + (size_t)(b * T_ + t0 + t) * D_ + d);
    ax += bt * cv.x;
    ay += bt * cv.y;
  }
  atomicAdd(&q2c[b * D_ + d], ax);
  atomicAdd(&q2c[b * D_ + d + 1], ay);
}

// ---------------- Kernel 5: out[..,1536:2048] = c * q2c (broadcast) ----------------
__global__ __launch_bounds__(128) void final_kernel(const float* __restrict__ c,
                                                    const float* __restrict__ q2c,
                                                    float* __restrict__ out) {
  int row = blockIdx.x;       // b*T_ + t
  int b = row >> 11;
  const float* cr = c + (size_t)row * D_;
  const float* qc = q2c + b * D_;
  float* o = out + (size_t)row * 2048 + 1536;
  int d = threadIdx.x * 4;
  float4 cv = *(const float4*)(cr + d);
  float4 qv = *(const float4*)(qc + d);
  float4 r = make_float4(cv.x * qv.x, cv.y * qv.y, cv.z * qv.z, cv.w * qv.w);
  *(float4*)(o + d) = r;
}

extern "C" void kernel_launch(void* const* d_in, const int* in_sizes, int n_in,
                              void* d_out, int out_size, void* d_ws, size_t ws_size,
                              hipStream_t stream) {
  (void)in_sizes; (void)n_in; (void)out_size; (void)ws_size;
  const float* c    = (const float*)d_in[0];
  const float* q    = (const float*)d_in[1];
  const float* w_c  = (const float*)d_in[2];
  const float* w_q  = (const float*)d_in[4];
  const float* w_cq = (const float*)d_in[6];
  // biases (d_in[3], d_in[5], d_in[7]) provably cancel in both softmaxes.
  float* out = (float*)d_out;
  float* ws   = (float*)d_ws;
  float* sq   = ws;                        // 8*512
  float* m    = ws + 4096;                 // 8*2048
  float* beta = ws + 4096 + 16384;         // 8*2048
  float* q2c  = ws + 4096 + 16384 + 16384; // 8*512

  sq_kernel<<<B_ * Q_ / 8, 256, 0, stream>>>(q, w_q, sq);
  attn_main<<<dim3(T_ / 16, B_), 256, 0, stream>>>(c, q, w_c, w_cq, sq, m, out);
  beta_kernel<<<B_, 256, 0, stream>>>(m, beta);
  hipMemsetAsync(q2c, 0, (size_t)B_ * D_ * sizeof(float), stream);
  q2c_kernel<<<dim3(T_ / 128, B_), 256, 0, stream>>>(c, beta, q2c);
  final_kernel<<<B_ * T_, 128, 0, stream>>>(c, q2c, out);
}
